// GMUCp_51754355916881
// MI455X (gfx1250) — matrix operand order repaired and checked by reading, not compiled, not run on hardware
//
#include <hip/hip_runtime.h>
#include <hip/hip_bf16.h>
#include <math.h>

// ---------------------------------------------------------------------------
// GMUC few-shot matcher on MI455X (gfx1250, wave32, WMMA f32_16x16x32_f16).
// All GEMMs run on v_wmma_f32_16x16x32_f16 with f32 accumulate; softmax /
// layernorm / LSTM nonlinearities stay f32.  Matchnet exploits softmax over a
// single support row == 1, so r == support (constant).
// ---------------------------------------------------------------------------

typedef __attribute__((ext_vector_type(16))) _Float16 v16h;
typedef __attribute__((ext_vector_type(8)))  float    v8f;

union HalfPack { v16h v; uint4 q[2]; unsigned int u[8]; _Float16 h[16]; };

#define WMMA_F16(A_, B_, C_) \
  __builtin_amdgcn_wmma_f32_16x16x32_f16(false, (A_), false, (B_), (short)0, (C_), false, false)

__device__ __forceinline__ float sigm(float x) { return 1.0f / (1.0f + expf(-x)); }

// A-fragment (16x32 f16) from an LDS tile with row pitch `pitch` halves.
// ISA 7.12.2: lanes 0-15 -> M=lane, kbase=0; lanes 16-31 -> M=lane-16, kbase=8.
// VGPR j<4 holds K pairs kbase+2j; j>=4 holds kbase+16+2(j-4).
__device__ __forceinline__ v16h afrag_lds(const _Float16* lds, int pitch, int k0) {
  int lane = threadIdx.x & 31;
  int m    = lane & 15;
  int kb   = (lane >> 4) << 3;            // 0 or 8
  const _Float16* p = lds + m * pitch + k0 + kb;
  HalfPack hp;
  hp.q[0] = *(const uint4*)(p);           // halves kb..kb+7
  hp.q[1] = *(const uint4*)(p + 16);      // halves kb+16..kb+23
  return hp.v;
}

// B-fragment (32x16 f16) from packed weights P[((k>>5)*N + n)*32 + (k&31)].
// Lanes 0-15: column n=lane, K=kt*32+{0..7,16..23} ; lanes 16-31: K=kt*32+{8..15,24..31}.
__device__ __forceinline__ v16h bfrag_packed(const _Float16* Bp, int N, int kt, int ncol0) {
  int lane = threadIdx.x & 31;
  int n    = ncol0 + (lane & 15);
  int hi   = lane >> 4;
  const _Float16* p = Bp + ((size_t)kt * N + n) * 32 + hi * 8;
  HalfPack hp;
  hp.q[0] = *(const uint4*)(p);
  hp.q[1] = *(const uint4*)(p + 16);
  return hp.v;
}

// ---------------------------------------------------------------------------
// Weight packer: W (N x K, row major f32) -> packed f16 [K/32][N][32]
// ---------------------------------------------------------------------------
__global__ void pack_w(const float* __restrict__ W, _Float16* __restrict__ P, int N, int K) {
  for (size_t idx = (size_t)blockIdx.x * blockDim.x + threadIdx.x; idx < (size_t)N * K;
       idx += (size_t)gridDim.x * blockDim.x) {
    int n = (int)(idx / K), k = (int)(idx % K);
    P[(((size_t)(k >> 5)) * N + n) * 32 + (k & 31)] = (_Float16)W[idx];
  }
}

__global__ void add_vec(const float* __restrict__ a, const float* __restrict__ b,
                        float* __restrict__ o, int n) {
  int i = blockIdx.x * blockDim.x + threadIdx.x;
  if (i < n) o[i] = a[i] + b[i];
}

__global__ void zero_f32(float* __restrict__ p, size_t n) {
  for (size_t i = (size_t)blockIdx.x * blockDim.x + threadIdx.x; i < n;
       i += (size_t)gridDim.x * blockDim.x) p[i] = 0.0f;
}

// ---------------------------------------------------------------------------
// Generic WMMA GEMM: C[M,N] = act( A[M,K](f16) * Bt[K,N](packed f16) + bias + Cin )
// block = 128 threads (4 waves); block tile 16 rows x 64 cols; A staged in LDS.
// ---------------------------------------------------------------------------
__global__ void __launch_bounds__(128)
gemm16(const _Float16* __restrict__ A, const _Float16* __restrict__ Bp,
       const float* __restrict__ bias, const float* __restrict__ Cin,
       float* __restrict__ Cf, _Float16* __restrict__ Ch,
       int M, int N, int K, int relu)
{
  extern __shared__ char smem_raw[];
  _Float16* lds = (_Float16*)smem_raw;         // 16 x (K+8)
  const int KP = K + 8;
  const int row0 = blockIdx.x * 16;

  // cooperative A-tile load (f16 -> LDS), zero out-of-range rows
  const int segs = K / 8;
  for (int idx = threadIdx.x; idx < 16 * segs; idx += 128) {
    int i = idx / segs, seg = idx % segs;
    uint4 val = make_uint4(0u, 0u, 0u, 0u);
    int row = row0 + i;
    if (row < M) val = *(const uint4*)(A + (size_t)row * K + seg * 8);
    *(uint4*)(lds + i * KP + seg * 8) = val;
  }
  __syncthreads();

  const int wave = threadIdx.x >> 5;
  const int lane = threadIdx.x & 31;
  const int ncol0 = blockIdx.y * 64 + wave * 16;
  const int n  = ncol0 + (lane & 15);
  const int hi = lane >> 4;

  float bz = bias ? bias[n] : 0.0f;
  v8f acc;
#pragma unroll
  for (int r = 0; r < 8; ++r) {
    float v = bz;
    if (Cin) {
      int row = row0 + r + 8 * hi;
      if (row < M) v += Cin[(size_t)row * N + n];
    }
    acc[r] = v;
  }

  for (int kt = 0; kt < K / 32; ++kt) {
    v16h a = afrag_lds(lds, KP, kt * 32);
    v16h b = bfrag_packed(Bp, N, kt, ncol0);
    acc = WMMA_F16(a, b, acc);
  }

#pragma unroll
  for (int r = 0; r < 8; ++r) {
    int row = row0 + r + 8 * hi;
    if (row >= M) continue;
    float v = acc[r];
    if (relu) v = v > 0.0f ? v : 0.0f;
    if (Cf) Cf[(size_t)row * N + n] = v;
    if (Ch) Ch[(size_t)row * N + n] = (_Float16)v;
  }
}

// ---------------------------------------------------------------------------
// Neighbor encoder stage 1: attention logits for both (mean,var) paths.
// One wave per 16 flattened (b,neighbor) rows.  Gathers [rel|ent] embedding
// rows into LDS (f16, pitch 520 halves to avoid bank conflicts), then
// score[row] = num_b + sum_e tanh( (emb @ W^T)[row,e] + b[e] ) * num_w[e].
// ---------------------------------------------------------------------------
__global__ void __launch_bounds__(32)
neigh_score(const int* __restrict__ conn, const float* __restrict__ emb_mean,
            const float* __restrict__ emb_var,
            const _Float16* __restrict__ Wm, const _Float16* __restrict__ Wv,
            const float* __restrict__ bm, const float* __restrict__ bv,
            const float* __restrict__ um, const float* __restrict__ uv,
            const float* __restrict__ umb, const float* __restrict__ uvb,
            float* __restrict__ score_m, float* __restrict__ score_v, int Mtot)
{
  __shared__ _Float16 cm[16 * 520];
  __shared__ _Float16 cv[16 * 520];
  const int lane  = threadIdx.x;
  const int tile0 = blockIdx.x * 16;

  for (int i = 0; i < 16; ++i) {
    int gr = tile0 + i;
    bool ok = gr < Mtot;
    int rel = ok ? conn[gr * 3 + 0] : 0;
    int ent = ok ? conn[gr * 3 + 1] : 0;
    const float* rm = emb_mean + (size_t)rel * 256;
    const float* em = emb_mean + (size_t)ent * 256;
    const float* rv = emb_var  + (size_t)rel * 256;
    const float* ev = emb_var  + (size_t)ent * 256;
    for (int j = lane; j < 256; j += 32) {
      cm[i * 520 + j]       = ok ? (_Float16)rm[j] : (_Float16)0.0f;
      cm[i * 520 + 256 + j] = ok ? (_Float16)em[j] : (_Float16)0.0f;
      cv[i * 520 + j]       = ok ? (_Float16)rv[j] : (_Float16)0.0f;
      cv[i * 520 + 256 + j] = ok ? (_Float16)ev[j] : (_Float16)0.0f;
    }
  }
  __syncthreads();

  const int hi = lane >> 4;
  for (int path = 0; path < 2; ++path) {
    const _Float16* W   = path ? Wv : Wm;
    const float* bvec   = path ? bv : bm;
    const float* u      = path ? uv : um;
    const _Float16* src = path ? cv : cm;
    float sp[8] = {0, 0, 0, 0, 0, 0, 0, 0};
    for (int nt = 0; nt < 16; ++nt) {
      int ncol = nt * 16 + (lane & 15);
      float bz = bvec[ncol];
      v8f acc = {bz, bz, bz, bz, bz, bz, bz, bz};
      for (int kt = 0; kt < 16; ++kt) {
        v16h a = afrag_lds(src, 520, kt * 32);
        v16h b = bfrag_packed(W, 256, kt, nt * 16);
        acc = WMMA_F16(a, b, acc);
      }
      float uw = u[ncol];
#pragma unroll
      for (int r = 0; r < 8; ++r) sp[r] += tanhf(acc[r]) * uw;
    }
    float nb = (path ? uvb : umb)[0];
    float* sc = path ? score_v : score_m;
#pragma unroll
    for (int r = 0; r < 8; ++r) {
      float v = sp[r];
      v += __shfl_xor(v, 1, 32);
      v += __shfl_xor(v, 2, 32);
      v += __shfl_xor(v, 4, 32);
      v += __shfl_xor(v, 8, 32);          // sum over the 16 lanes of each half
      int gr = tile0 + r + 8 * hi;
      if ((lane & 15) == 0 && gr < Mtot) sc[gr] = v + nb;
    }
  }
}

// ---------------------------------------------------------------------------
// Neighbor encoder stage 2: softmax over NB=30, out = tanh(sum a_n * eem_n).
// Note: both mean and var paths weight the SAME eem (per the reference).
// Writes into x buffers (f32 + f16) at column offset (left=0 / right=256).
// ---------------------------------------------------------------------------
__global__ void __launch_bounds__(256)
neigh_out(const int* __restrict__ conn, const float* __restrict__ emb_mean,
          const float* __restrict__ score_m, const float* __restrict__ score_v,
          float* __restrict__ xm32, float* __restrict__ xv32,
          _Float16* __restrict__ xm16, _Float16* __restrict__ xv16, int colOff)
{
  const int b = blockIdx.x;
  __shared__ float am[32], av[32];
  __shared__ int ents[32];
  const int t = threadIdx.x;
  if (t < 30) {
    am[t] = score_m[b * 30 + t];
    av[t] = score_v[b * 30 + t];
    ents[t] = conn[(b * 30 + t) * 3 + 1];
  }
  __syncthreads();
  if (t == 0) {
    float mx = -1e30f;
    for (int n2 = 0; n2 < 30; ++n2) mx = fmaxf(mx, am[n2]);
    float s = 0.0f;
    for (int n2 = 0; n2 < 30; ++n2) { am[n2] = expf(am[n2] - mx); s += am[n2]; }
    for (int n2 = 0; n2 < 30; ++n2) am[n2] /= s;
  }
  if (t == 1) {
    float mx = -1e30f;
    for (int n2 = 0; n2 < 30; ++n2) mx = fmaxf(mx, av[n2]);
    float s = 0.0f;
    for (int n2 = 0; n2 < 30; ++n2) { av[n2] = expf(av[n2] - mx); s += av[n2]; }
    for (int n2 = 0; n2 < 30; ++n2) av[n2] /= s;
  }
  __syncthreads();
  float accm = 0.0f, accv = 0.0f;
  for (int n2 = 0; n2 < 30; ++n2) {
    float v = emb_mean[(size_t)ents[n2] * 256 + t];
    accm += am[n2] * v;
    accv += av[n2] * v;
  }
  float om = tanhf(accm), ov = tanhf(accv);
  size_t o = (size_t)b * 512 + colOff + t;
  xm32[o] = om;  xv32[o] = ov;
  xm16[o] = (_Float16)om;  xv16[o] = (_Float16)ov;
}

// ---------------------------------------------------------------------------
// Residual LayerNorm: y = (z-mu)/(std_ddof1 + 1e-3)*a + g, z = o + x, D=512.
// ---------------------------------------------------------------------------
__global__ void __launch_bounds__(256)
ln_kernel(const float* __restrict__ o, const float* __restrict__ x,
          const float* __restrict__ a, const float* __restrict__ g,
          float* __restrict__ of32, _Float16* __restrict__ of16)
{
  const int row = blockIdx.x, t = threadIdx.x;
  __shared__ float red[256];
  __shared__ float mu_s, sd_s;
  const float* op = o + (size_t)row * 512;
  const float* xp = x + (size_t)row * 512;
  float z0 = op[t] + xp[t];
  float z1 = op[t + 256] + xp[t + 256];
  red[t] = z0 + z1;
  __syncthreads();
  for (int s = 128; s > 0; s >>= 1) { if (t < s) red[t] += red[t + s]; __syncthreads(); }
  if (t == 0) mu_s = red[0] * (1.0f / 512.0f);
  __syncthreads();
  float mu = mu_s;
  float d0 = z0 - mu, d1 = z1 - mu;
  red[t] = d0 * d0 + d1 * d1;
  __syncthreads();
  for (int s = 128; s > 0; s >>= 1) { if (t < s) red[t] += red[t + s]; __syncthreads(); }
  if (t == 0) sd_s = sqrtf(red[0] * (1.0f / 511.0f));
  __syncthreads();
  float inv = 1.0f / (sd_s + 0.001f);
  float y0 = d0 * inv * a[t] + g[t];
  float y1 = d1 * inv * a[t + 256] + g[t + 256];
  size_t base = (size_t)row * 512;
  of32[base + t] = y0;
  of32[base + t + 256] = y1;
  if (of16) { of16[base + t] = (_Float16)y0; of16[base + t + 256] = (_Float16)y1; }
}

// ---------------------------------------------------------------------------
// Matchnet LSTM elementwise: from gates g (B x 4096) update c, h, h_r=[h,s].
// ---------------------------------------------------------------------------
__global__ void __launch_bounds__(256)
lstm_elem(const float* __restrict__ g, float* __restrict__ c,
          const float* __restrict__ query, const float* __restrict__ s,
          float* __restrict__ hf32, _Float16* __restrict__ hr16, int B)
{
  size_t idx = (size_t)blockIdx.x * blockDim.x + threadIdx.x;
  if (idx >= (size_t)B * 1024) return;
  size_t b = idx >> 10;
  int j = (int)(idx & 1023);
  const float* gb = g + b * 4096;
  float gi = gb[j], gfv = gb[1024 + j], gg = gb[2048 + j], go = gb[3072 + j];
  float cn = sigm(gfv) * c[idx] + sigm(gi) * tanhf(gg);
  c[idx] = cn;
  float hr = sigm(go) * tanhf(cn);
  if (j < 512) {
    float h = query[b * 512 + j] + hr;
    hf32[b * 512 + j] = h;
    hr16[b * 1024 + j] = (_Float16)h;
  } else {
    hr16[b * 1024 + j] = (_Float16)s[j - 512];   // r == support (softmax of one)
  }
}

__global__ void __launch_bounds__(256)
dot512(const float* __restrict__ h, const float* __restrict__ s, float* __restrict__ out)
{
  const int b = blockIdx.x, t = threadIdx.x;
  __shared__ float red[256];
  red[t] = h[(size_t)b * 512 + t] * s[t] + h[(size_t)b * 512 + 256 + t] * s[256 + t];
  __syncthreads();
  for (int st = 128; st > 0; st >>= 1) { if (t < st) red[t] += red[t + st]; __syncthreads(); }
  if (t == 0) out[b] = red[0];
}

__global__ void finalize_k(const float* __restrict__ ms, const float* __restrict__ mv,
                           const float* __restrict__ w, const float* __restrict__ bb,
                           float* __restrict__ outp, int B)
{
  int i = blockIdx.x * blockDim.x + threadIdx.x;
  if (i < B) {
    float conf = 1.0f / (1.0f + expf(-(w[0] * mv[i] + bb[0])));
    outp[i] = ms[i] * conf;        // rank_score
    outp[B + i] = conf;            // conf_score
  }
}

// ---------------------------------------------------------------------------
// Set encoder (tiny, sequential): 5-step enc/dec LSTM on sgm (5x512), set
// attention -> sm (512); attention-only var path on sgv -> sv (512).
// One block, 1024 threads, all-f32 in LDS.
// ---------------------------------------------------------------------------
__global__ void __launch_bounds__(1024)
set_encoder(const float* __restrict__ sgm, const float* __restrict__ sgv,
            const float* __restrict__ eWih, const float* __restrict__ ebih,
            const float* __restrict__ eWhh, const float* __restrict__ ebhh,
            const float* __restrict__ dWih, const float* __restrict__ dbih,
            const float* __restrict__ dWhh, const float* __restrict__ dbhh,
            const float* __restrict__ sWm, const float* __restrict__ sWm_b,
            const float* __restrict__ sum_w, const float* __restrict__ sum_b,
            const float* __restrict__ sWv, const float* __restrict__ sWv_b,
            const float* __restrict__ suv_w, const float* __restrict__ suv_b,
            float* __restrict__ sm_out, float* __restrict__ sv_out)
{
  __shared__ float sg[5][512];     // sgm, later enc, later sgv
  __shared__ float sg0[5][512];    // enc_out, later enc+dec_out
  __shared__ float h[512], c[512], hf[512], cf[512], xv[512];
  __shared__ float gbuf[2048];
  __shared__ float sc[8];
  const int t = threadIdx.x;

  for (int i = t; i < 5 * 512; i += 1024) sg[i >> 9][i & 511] = sgm[i];
  if (t < 512) { h[t] = 0.0f; c[t] = 0.0f; }
  __syncthreads();

  // encoder scan
  for (int step = 0; step < 5; ++step) {
    for (int jj = t; jj < 2048; jj += 1024) {
      const float* wi = eWih + (size_t)jj * 512;
      const float* wh = eWhh + (size_t)jj * 512;
      float acc = ebih[jj] + ebhh[jj];
      for (int k = 0; k < 512; ++k) acc += wi[k] * sg[step][k] + wh[k] * h[k];
      gbuf[jj] = acc;
    }
    __syncthreads();
    if (t < 512) {
      float cn = sigm(gbuf[512 + t]) * c[t] + sigm(gbuf[t]) * tanhf(gbuf[1024 + t]);
      c[t] = cn;
      h[t] = sigm(gbuf[1536 + t]) * tanhf(cn);
      sg0[step][t] = h[t];                 // enc_out
    }
    __syncthreads();
  }
  // enc = sg + enc_out ; save final (hf, cf); x = enc[4]
  for (int i = t; i < 5 * 512; i += 1024) { int s_ = i >> 9, d = i & 511; sg[s_][d] += sg0[s_][d]; }
  if (t < 512) { hf[t] = h[t]; cf[t] = c[t]; }
  __syncthreads();
  if (t < 512) { xv[t] = sg[4][t]; h[t] = hf[t]; c[t] = cf[t]; }
  __syncthreads();

  // decoder scan: (x,h,c) -> cell -> carry x = new h ; sg0 = enc + dec_out
  for (int step = 0; step < 5; ++step) {
    for (int jj = t; jj < 2048; jj += 1024) {
      const float* wi = dWih + (size_t)jj * 512;
      const float* wh = dWhh + (size_t)jj * 512;
      float acc = dbih[jj] + dbhh[jj];
      for (int k = 0; k < 512; ++k) acc += wi[k] * xv[k] + wh[k] * h[k];
      gbuf[jj] = acc;
    }
    __syncthreads();
    if (t < 512) {
      float cn = sigm(gbuf[512 + t]) * c[t] + sigm(gbuf[t]) * tanhf(gbuf[1024 + t]);
      c[t] = cn;
      float hn = sigm(gbuf[1536 + t]) * tanhf(cn);
      h[t] = hn;
      sg0[step][t] = sg[step][t] + hn;
    }
    __syncthreads();
    if (t < 512) xv[t] = h[t];
    __syncthreads();
  }

  // mean-path set attention over sg0
  if (t < 8) sc[t] = 0.0f;
  __syncthreads();
  for (int i = t; i < 5 * 256; i += 1024) {
    int s_ = i >> 8, e = i & 255;
    const float* w = sWm + (size_t)e * 512;
    float acc = sWm_b[e];
    for (int k = 0; k < 512; ++k) acc += w[k] * sg0[s_][k];
    atomicAdd(&sc[s_], tanhf(acc) * sum_w[e]);
  }
  __syncthreads();
  if (t == 0) {
    float mx = -1e30f;
    for (int s_ = 0; s_ < 5; ++s_) { sc[s_] += sum_b[0]; mx = fmaxf(mx, sc[s_]); }
    float ssum = 0.0f;
    for (int s_ = 0; s_ < 5; ++s_) { sc[s_] = expf(sc[s_] - mx); ssum += sc[s_]; }
    for (int s_ = 0; s_ < 5; ++s_) sc[s_] /= ssum;
  }
  __syncthreads();
  if (t < 512) {
    float acc = 0.0f;
    for (int s_ = 0; s_ < 5; ++s_) acc += sc[s_] * sg0[s_][t];
    sm_out[t] = acc;
  }
  __syncthreads();

  // var-path attention directly over sgv
  for (int i = t; i < 5 * 512; i += 1024) sg[i >> 9][i & 511] = sgv[i];
  if (t < 8) sc[t] = 0.0f;
  __syncthreads();
  for (int i = t; i < 5 * 256; i += 1024) {
    int s_ = i >> 8, e = i & 255;
    const float* w = sWv + (size_t)e * 512;
    float acc = sWv_b[e];
    for (int k = 0; k < 512; ++k) acc += w[k] * sg[s_][k];
    atomicAdd(&sc[s_], tanhf(acc) * suv_w[e]);
  }
  __syncthreads();
  if (t == 0) {
    float mx = -1e30f;
    for (int s_ = 0; s_ < 5; ++s_) { sc[s_] += suv_b[0]; mx = fmaxf(mx, sc[s_]); }
    float ssum = 0.0f;
    for (int s_ = 0; s_ < 5; ++s_) { sc[s_] = expf(sc[s_] - mx); ssum += sc[s_]; }
    for (int s_ = 0; s_ < 5; ++s_) sc[s_] /= ssum;
  }
  __syncthreads();
  if (t < 512) {
    float acc = 0.0f;
    for (int s_ = 0; s_ < 5; ++s_) acc += sc[s_] * sg[s_][t];
    sv_out[t] = acc;
  }
}

// ---------------------------------------------------------------------------
// Host orchestration
// ---------------------------------------------------------------------------
extern "C" void kernel_launch(void* const* d_in, const int* in_sizes, int n_in,
                              void* d_out, int out_size, void* d_ws, size_t ws_size,
                              hipStream_t stream)
{
  (void)in_sizes; (void)n_in; (void)out_size; (void)ws_size;
  const int B = 2048, NB = 30, FEW = 5;

  // params flattened in setup_inputs()/dict insertion order
  const float* emb_mean = (const float*)d_in[0];
  const float* emb_var  = (const float*)d_in[1];
  const float* nWm_w = (const float*)d_in[2];
  const float* nWm_b = (const float*)d_in[3];
  const float* num_w = (const float*)d_in[4];
  const float* num_b = (const float*)d_in[5];
  const float* nWv_w = (const float*)d_in[6];
  const float* nWv_b = (const float*)d_in[7];
  const float* nuv_w = (const float*)d_in[8];
  const float* nuv_b = (const float*)d_in[9];
  const float* enc_Wih = (const float*)d_in[10];
  const float* enc_bih = (const float*)d_in[11];
  const float* enc_Whh = (const float*)d_in[12];
  const float* enc_bhh = (const float*)d_in[13];
  const float* dec_Wih = (const float*)d_in[14];
  const float* dec_bih = (const float*)d_in[15];
  const float* dec_Whh = (const float*)d_in[16];
  const float* dec_bhh = (const float*)d_in[17];
  const float* sWm_w = (const float*)d_in[18];
  const float* sWm_b = (const float*)d_in[19];
  const float* sum_w = (const float*)d_in[20];
  const float* sum_b = (const float*)d_in[21];
  const float* sWv_w = (const float*)d_in[22];
  const float* sWv_b = (const float*)d_in[23];
  const float* suv_w = (const float*)d_in[24];
  const float* suv_b = (const float*)d_in[25];
  const float* sem_p1w = (const float*)d_in[26];
  const float* sem_p1b = (const float*)d_in[27];
  const float* sem_p2w = (const float*)d_in[28];
  const float* sem_p2b = (const float*)d_in[29];
  const float* sem_a = (const float*)d_in[30];
  const float* sem_g = (const float*)d_in[31];
  const float* mnm_Wih = (const float*)d_in[32];
  const float* mnm_bih = (const float*)d_in[33];
  const float* mnm_Whh = (const float*)d_in[34];
  const float* mnm_bhh = (const float*)d_in[35];
  const float* sev_p1w = (const float*)d_in[36];
  const float* sev_p1b = (const float*)d_in[37];
  const float* sev_p2w = (const float*)d_in[38];
  const float* sev_p2b = (const float*)d_in[39];
  const float* sev_a = (const float*)d_in[40];
  const float* sev_g = (const float*)d_in[41];
  const float* mnv_Wih = (const float*)d_in[42];
  const float* mnv_bih = (const float*)d_in[43];
  const float* mnv_Whh = (const float*)d_in[44];
  const float* mnv_bhh = (const float*)d_in[45];
  const float* p_w = (const float*)d_in[46];
  const float* p_b = (const float*)d_in[47];
  // 48 = support (unused), degrees unused
  const int* sl_conn = (const int*)d_in[49];
  const int* sr_conn = (const int*)d_in[51];
  const int* ql_conn = (const int*)d_in[53];
  const int* qr_conn = (const int*)d_in[55];

  // ---- workspace bump allocator (aligned 256B) ----
  char* ws = (char*)d_ws;
  size_t off = 0;
  auto alloc = [&](size_t bytes) -> void* {
    void* p = ws + off;
    off = (off + bytes + 255) & ~(size_t)255;
    return p;
  };
  _Float16* pk_nWm   = (_Float16*)alloc((size_t)512 * 256 * 2);
  _Float16* pk_nWv   = (_Float16*)alloc((size_t)512 * 256 * 2);
  _Float16* pk_semp1 = (_Float16*)alloc((size_t)512 * 1024 * 2);
  _Float16* pk_semp2 = (_Float16*)alloc((size_t)1024 * 512 * 2);
  _Float16* pk_sevp1 = (_Float16*)alloc((size_t)512 * 1024 * 2);
  _Float16* pk_sevp2 = (_Float16*)alloc((size_t)1024 * 512 * 2);
  _Float16* pk_mnm_ih = (_Float16*)alloc((size_t)512 * 4096 * 2);
  _Float16* pk_mnm_hh = (_Float16*)alloc((size_t)1024 * 4096 * 2);
  _Float16* pk_mnv_ih = (_Float16*)alloc((size_t)512 * 4096 * 2);
  _Float16* pk_mnv_hh = (_Float16*)alloc((size_t)1024 * 4096 * 2);
  float* mnm_bias = (float*)alloc(4096 * 4);
  float* mnv_bias = (float*)alloc(4096 * 4);
  float* sc_m = (float*)alloc((size_t)B * NB * 4);
  float* sc_v = (float*)alloc((size_t)B * NB * 4);
  float* xq_m = (float*)alloc((size_t)B * 512 * 4);
  float* xq_v = (float*)alloc((size_t)B * 512 * 4);
  _Float16* xq_m_h = (_Float16*)alloc((size_t)B * 512 * 2);
  _Float16* xq_v_h = (_Float16*)alloc((size_t)B * 512 * 2);
  float* xs_m = (float*)alloc((size_t)FEW * 512 * 4);
  float* xs_v = (float*)alloc((size_t)FEW * 512 * 4);
  _Float16* xs_m_h = (_Float16*)alloc((size_t)FEW * 512 * 2);
  _Float16* xs_v_h = (_Float16*)alloc((size_t)FEW * 512 * 2);
  _Float16* h_se = (_Float16*)alloc((size_t)B * 1024 * 2);
  float* o_se = (float*)alloc((size_t)B * 512 * 4);
  float* qgm = (float*)alloc((size_t)B * 512 * 4);
  float* qgv = (float*)alloc((size_t)B * 512 * 4);
  _Float16* qgm_h = (_Float16*)alloc((size_t)B * 512 * 2);
  _Float16* qgv_h = (_Float16*)alloc((size_t)B * 512 * 2);
  float* sgm = (float*)alloc((size_t)FEW * 512 * 4);
  float* sgv = (float*)alloc((size_t)FEW * 512 * 4);
  float* smv_m = (float*)alloc(512 * 4);
  float* smv_v = (float*)alloc(512 * 4);
  float* gbuf = (float*)alloc((size_t)B * 4096 * 4);
  float* cbuf = (float*)alloc((size_t)B * 1024 * 4);
  float* hbuf = (float*)alloc((size_t)B * 512 * 4);
  _Float16* hr_h = (_Float16*)alloc((size_t)B * 1024 * 2);
  float* ms = (float*)alloc(B * 4);
  float* mvv = (float*)alloc(B * 4);

  auto pack = [&](const float* W, _Float16* P, int N, int K) {
    pack_w<<<1024, 256, 0, stream>>>(W, P, N, K);
  };
  auto gemm = [&](const _Float16* A, const _Float16* Bp, const float* bias, const float* Cin,
                  float* Cf, _Float16* Ch, int M, int N, int K, int relu) {
    dim3 grid((M + 15) / 16, N / 64);
    size_t sh = (size_t)16 * (K + 8) * sizeof(_Float16);
    gemm16<<<grid, 128, sh, stream>>>(A, Bp, bias, Cin, Cf, Ch, M, N, K, relu);
  };

  // ---- 1. pack weights to f16 (WMMA-B layout), combine matchnet biases ----
  pack(nWm_w, pk_nWm, 256, 512);
  pack(nWv_w, pk_nWv, 256, 512);
  pack(sem_p1w, pk_semp1, 1024, 512);
  pack(sem_p2w, pk_semp2, 512, 1024);
  pack(sev_p1w, pk_sevp1, 1024, 512);
  pack(sev_p2w, pk_sevp2, 512, 1024);
  pack(mnm_Wih, pk_mnm_ih, 4096, 512);
  pack(mnm_Whh, pk_mnm_hh, 4096, 1024);
  pack(mnv_Wih, pk_mnv_ih, 4096, 512);
  pack(mnv_Whh, pk_mnv_hh, 4096, 1024);
  add_vec<<<16, 256, 0, stream>>>(mnm_bih, mnm_bhh, mnm_bias, 4096);
  add_vec<<<16, 256, 0, stream>>>(mnv_bih, mnv_bhh, mnv_bias, 4096);

  // ---- 2. neighbor encoders (4 connection sets) ----
  struct Set { const int* conn; int rows; float* xm; float* xv; _Float16* xmh; _Float16* xvh; int colOff; };
  Set sets[4] = {
    { ql_conn, B,   xq_m, xq_v, xq_m_h, xq_v_h, 0   },
    { qr_conn, B,   xq_m, xq_v, xq_m_h, xq_v_h, 256 },
    { sl_conn, FEW, xs_m, xs_v, xs_m_h, xs_v_h, 0   },
    { sr_conn, FEW, xs_m, xs_v, xs_m_h, xs_v_h, 256 },
  };
  for (int s = 0; s < 4; ++s) {
    int Mtot = sets[s].rows * NB;
    neigh_score<<<(Mtot + 15) / 16, 32, 0, stream>>>(
        sets[s].conn, emb_mean, emb_var, pk_nWm, pk_nWv, nWm_b, nWv_b,
        num_w, nuv_w, num_b, nuv_b, sc_m, sc_v, Mtot);
    neigh_out<<<sets[s].rows, 256, 0, stream>>>(
        sets[s].conn, emb_mean, sc_m, sc_v,
        sets[s].xm, sets[s].xv, sets[s].xmh, sets[s].xvh, sets[s].colOff);
  }

  // ---- 3. support encoders (residual MLP + LN) ----
  struct SE { const _Float16* xh; const float* x32; const _Float16* p1; const float* p1b;
              const _Float16* p2; const float* p2b; const float* a; const float* g;
              float* out32; _Float16* out16; int M; };
  SE ses[4] = {
    { xq_m_h, xq_m, pk_semp1, sem_p1b, pk_semp2, sem_p2b, sem_a, sem_g, qgm, qgm_h, B   },
    { xq_v_h, xq_v, pk_sevp1, sev_p1b, pk_sevp2, sev_p2b, sev_a, sev_g, qgv, qgv_h, B   },
    { xs_m_h, xs_m, pk_semp1, sem_p1b, pk_semp2, sem_p2b, sem_a, sem_g, sgm, nullptr, FEW },
    { xs_v_h, xs_v, pk_sevp1, sev_p1b, pk_sevp2, sev_p2b, sev_a, sev_g, sgv, nullptr, FEW },
  };
  for (int s = 0; s < 4; ++s) {
    gemm(ses[s].xh, ses[s].p1, ses[s].p1b, nullptr, nullptr, h_se, ses[s].M, 1024, 512, 1);
    gemm(h_se, ses[s].p2, ses[s].p2b, nullptr, o_se, nullptr, ses[s].M, 512, 1024, 0);
    ln_kernel<<<ses[s].M, 256, 0, stream>>>(o_se, ses[s].x32, ses[s].a, ses[s].g,
                                            ses[s].out32, ses[s].out16);
  }

  // ---- 4. set encoder (tiny sequential) -> sm, sv ----
  set_encoder<<<1, 1024, 0, stream>>>(
      sgm, sgv, enc_Wih, enc_bih, enc_Whh, enc_bhh, dec_Wih, dec_bih, dec_Whh, dec_bhh,
      sWm_w, sWm_b, sum_w, sum_b, sWv_w, sWv_b, suv_w, suv_b, smv_m, smv_v);

  // ---- 5. matchnet, tags m then v ----
  struct MN { const _Float16* qh; const float* q32; const _Float16* Wih; const _Float16* Whh;
              const float* bias; const float* s; float* out; };
  MN mns[2] = {
    { qgm_h, qgm, pk_mnm_ih, pk_mnm_hh, mnm_bias, smv_m, ms  },
    { qgv_h, qgv, pk_mnv_ih, pk_mnv_hh, mnv_bias, smv_v, mvv },
  };
  const size_t lstmN = (size_t)B * 1024;
  for (int tg = 0; tg < 2; ++tg) {
    zero_f32<<<1024, 256, 0, stream>>>(cbuf, lstmN);
    zero_f32<<<1024, 256, 0, stream>>>((float*)hr_h, lstmN / 2);
    for (int step = 0; step < 4; ++step) {
      gemm(mns[tg].qh, mns[tg].Wih, mns[tg].bias, nullptr, gbuf, nullptr, B, 4096, 512, 0);
      gemm(hr_h, mns[tg].Whh, nullptr, gbuf, gbuf, nullptr, B, 4096, 1024, 0);
      lstm_elem<<<(unsigned)((lstmN + 255) / 256), 256, 0, stream>>>(
          gbuf, cbuf, mns[tg].q32, mns[tg].s, hbuf, hr_h, B);
    }
    dot512<<<B, 256, 0, stream>>>(hbuf, mns[tg].s, mns[tg].out);
  }

  // ---- 6. output: rank = ms * sigmoid(w*mv + b); conf ----
  finalize_k<<<(B + 255) / 256, 256, 0, stream>>>(ms, mvv, p_w, p_b, (float*)d_out, B);
}
